// CodebookEMA_40072044871898
// MI455X (gfx1250) — matrix operand order repaired and checked by reading, not compiled, not run on hardware
//
#include <hip/hip_runtime.h>
#include <hip/hip_bf16.h>
#include <stdint.h>

// ---------------------------------------------------------------------------
// CodebookEMA forward for MI455X (gfx1250, wave32, WMMA).
// Distance GEMM: v_wmma_f32_16x16x32_bf16 with bf16x3 split-precision fp32
// emulation; B tiles staged cooperatively into LDS via gfx1250 async copy
// (global_load_async_to_lds_b128 / s_wait_asynccnt), double-buffered so the
// next tile's DMA overlaps the current tile's 24 WMMAs. Memory-bound overall.
// ---------------------------------------------------------------------------

typedef __attribute__((ext_vector_type(16))) __bf16 v16bf;
typedef __attribute__((ext_vector_type(8)))  float  v8f;

#define DK_B   16
#define DK_T   4096
#define DK_D   256
#define DK_K   1024
#define DK_BT  65536
#define DECAYF 0.99f
#define EPSF   1e-5f

// workspace byte offsets
#define WS_PACKHI 0u        // 512 KB : packed B hi tiles (bf16, WMMA layout)
#define WS_PACKLO 524288u   // 512 KB : packed B lo tiles
#define WS_ENORM  1048576u  //   4 KB : ||e_k||^2
#define WS_COUNTS 1052672u  //   4 KB : counts[K]            (zeroed)
#define WS_EMBSUM 1056768u  //   1 MB : emb_sum[D,K]         (zeroed, contiguous)
#define WS_LOSS   2105344u  //  64 B  : per-batch loss accum (zeroed, contiguous)
#define WS_CSNORM 2105600u  //   4 KB : Laplace-normalized cluster size
#define WS_EMBT   2109696u  //   1 MB : new embedding transposed [K,D]
#define WS_IDX    3158272u  // 256 KB : argmin code per row

union BV16 { v16bf v; unsigned short u[16]; };

// gfx1250 async global->LDS copy, 16B per lane, tracked by ASYNCcnt.
// LDS address comes from the vdst VGPR; LDS offset == low 32 bits of generic
// shared-pointer (LDS aperture maps addr[31:0] to the LDS offset).
__device__ __forceinline__ void async_g2lds_b128(void* lds_ptr, const void* gptr) {
  unsigned lds_off = (unsigned)(uintptr_t)lds_ptr;
  asm volatile("global_load_async_to_lds_b128 %0, %1, off"
               :: "v"(lds_off), "v"(gptr)
               : "memory");
}
__device__ __forceinline__ void wait_asynccnt0() {
  asm volatile("s_wait_asynccnt 0x0" ::: "memory");
}

// ---------------- prep: ||e_k||^2 (exact f32) ------------------------------
__global__ void k_enorm(const float* __restrict__ emb, float* __restrict__ enorm) {
  int k = blockIdx.x * blockDim.x + threadIdx.x;
  if (k >= DK_K) return;
  float s = 0.f;
  for (int d = 0; d < DK_D; ++d) { float e = emb[d * DK_K + k]; s += e * e; }
  enorm[k] = s;
}

// ---------------- prep: pack codebook into WMMA-B bf16 hi/lo tiles ---------
// B tile (32x16 bf16): lanes 0-15 hold K=d0+0..15 (col N=lane), lanes 16-31
// hold K=d0+16..31 (col N=lane-16); 2 values per VGPR, sequential.
__global__ void k_pack(const float* __restrict__ emb,
                       v16bf* __restrict__ phi, v16bf* __restrict__ plo) {
  int wave = threadIdx.x >> 5, lane = threadIdx.x & 31;
  int chunk = blockIdx.x * 8 + wave;              // 0..511 = ct*8 + dc
  int ct = chunk >> 3, dc = chunk & 7;
  int d0 = dc * 32 + ((lane < 16) ? 0 : 16);
  int kc = ct * 16 + (lane & 15);
  BV16 H, L;
#pragma unroll
  for (int i = 0; i < 16; ++i) {
    float x = emb[(d0 + i) * DK_K + kc];
    unsigned xb = __float_as_uint(x);
    H.u[i] = (unsigned short)(xb >> 16);                         // truncate -> hi
    float rem = x - __uint_as_float(xb & 0xFFFF0000u);
    L.u[i] = (unsigned short)(__float_as_uint(rem) >> 16);       // residual -> lo
  }
  phi[chunk * 32 + lane] = H.v;
  plo[chunk * 32 + lane] = L.v;
}

__global__ void k_zero(float* p, int n) {
  int i = blockIdx.x * blockDim.x + threadIdx.x;
  if (i < n) p[i] = 0.f;
}

// ---------------- fused distance-GEMM + argmin -----------------------------
// One wave per 16-row tile (A 16x256 hi/lo in registers, reused across all
// 64 column tiles). B tiles (16KB hi+lo per column tile) are async-DMA'd
// into a double-buffered LDS stage once per workgroup and shared by 8 waves.
__global__ void __launch_bounds__(256)
k_gemm_argmin(const float* __restrict__ z,
              const char* __restrict__ phi, const char* __restrict__ plo,
              const float* __restrict__ enorm, int* __restrict__ idx) {
  __shared__ __align__(16) unsigned char sbuf[2][16384];  // [buf][hi 8KB | lo 8KB]

  const int t    = threadIdx.x;            // 0..255
  const int wave = t >> 5, lane = t & 31;
  const int rowBase = blockIdx.x * 128 + wave * 16;
  const int r = rowBase + (lane & 15);
  const int db = (lane < 16) ? 0 : 16;     // A layout: lanes<16 K in {0-15}
  const float* zr = z + (size_t)r * DK_D;

  // ---- build A tiles (hi/lo bf16, WMMA A layout) in registers ----
  v16bf ahi[8], alo[8];
#pragma unroll
  for (int dc = 0; dc < 8; ++dc) {
    BV16 H, L;
    const int d0 = dc * 32 + db;
#pragma unroll
    for (int i = 0; i < 8; ++i) {
      float x = zr[d0 + i];
      unsigned xb = __float_as_uint(x);
      H.u[i] = (unsigned short)(xb >> 16);
      float rx = x - __uint_as_float(xb & 0xFFFF0000u);
      L.u[i] = (unsigned short)(__float_as_uint(rx) >> 16);
      float y = zr[d0 + 8 + i];
      unsigned yb = __float_as_uint(y);
      H.u[8 + i] = (unsigned short)(yb >> 16);
      float ry = y - __uint_as_float(yb & 0xFFFF0000u);
      L.u[8 + i] = (unsigned short)(__float_as_uint(ry) >> 16);
    }
    ahi[dc] = H.v; alo[dc] = L.v;
  }

  // ---- stage helper: 16KB (8KB hi + 8KB lo) for one column tile ----
  auto stage = [&](int ct, int buf) {
    const char* gh = phi + (size_t)ct * 8192;
    const char* gl = plo + (size_t)ct * 8192;
    unsigned char* d = sbuf[buf];
#pragma unroll
    for (int i = 0; i < 2; ++i) {
      int e = (t + i * 256) * 16;                       // 2x256x16B = 8KB
      async_g2lds_b128(d + e,        gh + e);
      async_g2lds_b128(d + 8192 + e, gl + e);
    }
  };

  float bs[8]; int bi[8];
#pragma unroll
  for (int j = 0; j < 8; ++j) { bs[j] = 3.4e38f; bi[j] = 0; }

  stage(0, 0);                                          // prologue DMA

#pragma unroll 1
  for (int ct = 0; ct < 64; ++ct) {
    wait_asynccnt0();                                   // my DMA done
    __syncthreads();                                    // all waves' DMA visible
    if (ct + 1 < 64) stage(ct + 1, (ct + 1) & 1);       // overlap next tile DMA

    const v16bf* sh = (const v16bf*)(sbuf[ct & 1]);
    const v16bf* sl = (const v16bf*)(sbuf[ct & 1] + 8192);
    v8f acc = {};
#pragma unroll
    for (int dc = 0; dc < 8; ++dc) {
      v16bf bhv = sh[dc * 32 + lane];                   // ds_load_b128 x2
      v16bf blv = sl[dc * 32 + lane];
      acc = __builtin_amdgcn_wmma_f32_16x16x32_bf16(false, ahi[dc], false, bhv,
                                                    (short)0, acc, false, false);
      acc = __builtin_amdgcn_wmma_f32_16x16x32_bf16(false, ahi[dc], false, blv,
                                                    (short)0, acc, false, false);
      acc = __builtin_amdgcn_wmma_f32_16x16x32_bf16(false, alo[dc], false, bhv,
                                                    (short)0, acc, false, false);
    }
    // score = ||e||^2 - 2 z.e ; C layout: VGPR j, lanes<16 -> M=j, N=lane;
    // lanes>=16 -> M=8+j, N=lane-16.
    const int kc = ct * 16 + (lane & 15);
    const float en = enorm[kc];
#pragma unroll
    for (int j = 0; j < 8; ++j) {
      float s = en - 2.0f * acc[j];
      if (s < bs[j]) { bs[j] = s; bi[j] = kc; }         // strict < keeps first idx
    }
    __syncthreads();                                    // buffer reads done
  }

  // min-reduce across the 16 lanes holding one row (tie -> smaller k)
#pragma unroll
  for (int j = 0; j < 8; ++j) {
    float s = bs[j]; int k = bi[j];
#pragma unroll
    for (int m = 1; m < 16; m <<= 1) {
      float os = __shfl_xor(s, m, 32);
      int   ok = __shfl_xor(k, m, 32);
      if (os < s || (os == s && ok < k)) { s = os; k = ok; }
    }
    bs[j] = s; bi[j] = k;
  }
  if ((lane & 15) == 0) {
    const int half = lane >> 4;
#pragma unroll
    for (int j = 0; j < 8; ++j) idx[rowBase + half * 8 + j] = bi[j];
  }
}

// ---------------- scatter: counts + emb_sum + code -------------------------
__global__ void k_scatter(const float* __restrict__ z, const int* __restrict__ idx,
                          float* __restrict__ counts, float* __restrict__ embsum,
                          float* __restrict__ code_out) {
  int wave = threadIdx.x >> 5, lane = threadIdx.x & 31;
  int row = blockIdx.x * 8 + wave;
  int k = idx[row];
  if (lane == 0) { code_out[row] = (float)k; atomicAdd(counts + k, 1.0f); }
  const float* zr = z + (size_t)row * DK_D;
#pragma unroll 4
  for (int d = lane; d < DK_D; d += 32)
    atomicAdd(embsum + d * DK_K + k, zr[d]);
}

// ---------------- EMA cluster size + Laplace smoothing ---------------------
__global__ void k_cluster(const float* __restrict__ cs_in, const float* __restrict__ counts,
                          float* __restrict__ ncs_out, float* __restrict__ csnorm) {
  __shared__ float red[DK_K];
  int k = threadIdx.x;
  float ncs = cs_in[k] * DECAYF + (1.0f - DECAYF) * counts[k];
  ncs_out[k] = ncs;
  red[k] = ncs;
  __syncthreads();
  for (int s = DK_K / 2; s > 0; s >>= 1) {
    if (k < s) red[k] += red[k + s];
    __syncthreads();
  }
  float n = red[0];
  csnorm[k] = (ncs + EPSF) / (n + (float)DK_K * EPSF) * n;
}

// ---------------- EMA mean + new embedding (+ transposed copy) -------------
__global__ void k_final_emb(const float* __restrict__ mean_in, const float* __restrict__ embsum,
                            const float* __restrict__ csnorm,
                            float* __restrict__ mean_out, float* __restrict__ emb_out,
                            float* __restrict__ embT) {
  int i = blockIdx.x * blockDim.x + threadIdx.x;   // over D*K
  if (i >= DK_D * DK_K) return;
  int d = i >> 10, k = i & (DK_K - 1);
  float nm = mean_in[i] * DECAYF + (1.0f - DECAYF) * embsum[i];
  mean_out[i] = nm;
  float ne = nm / csnorm[k];
  emb_out[i] = ne;
  embT[k * DK_D + d] = ne;
}

// ---------------- gather z_q + commitment loss -----------------------------
__global__ void k_gather_loss(const float* __restrict__ z, const int* __restrict__ idx,
                              const float* __restrict__ embT,
                              float* __restrict__ zq, float* __restrict__ loss) {
  int wave = threadIdx.x >> 5, lane = threadIdx.x & 31;
  int row = blockIdx.x * 8 + wave;
  int k = idx[row];
  const float* et = embT + (size_t)k * DK_D;
  const float* zr = z + (size_t)row * DK_D;
  float* qo = zq + (size_t)row * DK_D;
  float ls = 0.f;
#pragma unroll 4
  for (int d = lane; d < DK_D; d += 32) {
    float q = et[d];
    qo[d] = q;                    // straight-through output == z_q numerically
    float df = q - zr[d];
    ls += df * df;
  }
#pragma unroll
  for (int m = 1; m < 32; m <<= 1) ls += __shfl_xor(ls, m, 32);
  if (lane == 0) atomicAdd(loss + (row >> 12), ls);   // b = row / T
}

__global__ void k_losses(const float* __restrict__ loss, float* __restrict__ commit,
                         float* __restrict__ codebook) {
  int b = threadIdx.x;
  if (b < DK_B) {
    commit[b] = loss[b] * (1.0f / (float)(DK_T * DK_D));
    codebook[b] = 0.f;
  }
}

// ---------------------------------------------------------------------------
extern "C" void kernel_launch(void* const* d_in, const int* in_sizes, int n_in,
                              void* d_out, int out_size, void* d_ws, size_t ws_size,
                              hipStream_t stream) {
  (void)in_sizes; (void)n_in; (void)out_size; (void)ws_size;
  const float* z     = (const float*)d_in[0];   // [B,T,D]
  const float* emb   = (const float*)d_in[1];   // [D,K]
  const float* cs    = (const float*)d_in[2];   // [K]
  const float* emean = (const float*)d_in[3];   // [D,K]

  char* ws = (char*)d_ws;
  v16bf* phi    = (v16bf*)(ws + WS_PACKHI);
  v16bf* plo    = (v16bf*)(ws + WS_PACKLO);
  float* enorm  = (float*)(ws + WS_ENORM);
  float* counts = (float*)(ws + WS_COUNTS);
  float* embsum = (float*)(ws + WS_EMBSUM);
  float* lossw  = (float*)(ws + WS_LOSS);
  float* csn    = (float*)(ws + WS_CSNORM);
  float* embT   = (float*)(ws + WS_EMBT);
  int*   idx    = (int*)  (ws + WS_IDX);

  float* o = (float*)d_out;                       // flat tuple in return order
  float* o_zq     = o;                            // [B,T,D]
  float* o_commit = o + 16777216;                 // [B]
  float* o_cb     = o + 16777232;                 // [B]
  float* o_code   = o + 16777248;                 // [B,T] (as float)
  float* o_emb    = o + 16842784;                 // [D,K]
  float* o_ncs    = o + 17104928;                 // [K]
  float* o_mean   = o + 17105952;                 // [D,K]

  k_enorm<<<4, 256, 0, stream>>>(emb, enorm);
  k_pack<<<64, 256, 0, stream>>>(emb, phi, plo);
  {
    int nz = DK_K + DK_D * DK_K + DK_B;           // counts+embsum+loss contiguous
    k_zero<<<(nz + 255) / 256, 256, 0, stream>>>(counts, nz);
  }
  k_gemm_argmin<<<512, 256, 0, stream>>>(z, (const char*)phi, (const char*)plo,
                                         enorm, idx);
  k_scatter<<<8192, 256, 0, stream>>>(z, idx, counts, embsum, o_code);
  k_cluster<<<1, DK_K, 0, stream>>>(cs, counts, o_ncs, csn);
  k_final_emb<<<1024, 256, 0, stream>>>(emean, embsum, csn, o_mean, o_emb, embT);
  k_gather_loss<<<8192, 256, 0, stream>>>(z, idx, embT, o_zq, lossw);
  k_losses<<<1, 64, 0, stream>>>(lossw, o_commit, o_cb);
}
